// StackedBidirectionalLstm_28355374088396
// MI455X (gfx1250) — matrix operand order repaired and checked by reading, not compiled, not run on hardware
//
#include <hip/hip_runtime.h>
#include <hip/hip_bf16.h>
#include <math.h>

// ---------------------------------------------------------------------------
// Types for CDNA5 WMMA (wave32): bf16 operands, f32 accumulate.
// ---------------------------------------------------------------------------
typedef __bf16 bf16;
typedef __attribute__((ext_vector_type(16))) __bf16 v16bf;
typedef __attribute__((ext_vector_type(8)))  __bf16 v8bf;
typedef __attribute__((ext_vector_type(8)))  float  v8f;

union AFrag { v16bf v; v8bf h[2]; };

#define Bsz   32
#define Tsz   512
#define Dsz   1024
#define Hsz   512
#define SIXH  3072
#define FIVEH 2560
#define MROWS (Bsz * Tsz)        // 16384
#define NUM_SCAN_WGS 32
#define SCAN_CW 16               // hidden columns owned per WG

static __device__ __forceinline__ float sigf(float x) {
    return 1.0f / (1.0f + __expf(-x));
}

// ---------------------------------------------------------------------------
// Utility kernels
// ---------------------------------------------------------------------------
__global__ void cvt_f32_to_bf16(const float* __restrict__ s,
                                bf16* __restrict__ d, size_t n) {
    size_t i  = (size_t)blockIdx.x * blockDim.x + threadIdx.x;
    size_t st = (size_t)gridDim.x * blockDim.x;
    for (; i < n; i += st) d[i] = (bf16)s[i];
}

__global__ void fill_f32(float* __restrict__ d, size_t n, float v) {
    size_t i  = (size_t)blockIdx.x * blockDim.x + threadIdx.x;
    size_t st = (size_t)gridDim.x * blockDim.x;
    for (; i < n; i += st) d[i] = v;
}

__global__ void fill_u32(unsigned int* __restrict__ d, int n) {
    int i = threadIdx.x;
    if (i < n) d[i] = 0u;
}

// ---------------------------------------------------------------------------
// Input-projection GEMM:  P[16384,3072] = A[16384,1024](bf16) * W[3072,1024]^T
// Register-tiled: each wave owns a 2(M) x 4(N) block of 16x16 tiles -> A frags
// reused over 4 N-tiles, B frags over 2 M-tiles: 8 WMMA per 12 b128 loads.
// grid = (16384/32, 3072/512) = (512, 6), block = 256 (8 waves).
// ---------------------------------------------------------------------------
__global__ __launch_bounds__(256) void gemm_pi(const bf16* __restrict__ A,
                                               const bf16* __restrict__ W,
                                               float* __restrict__ P) {
    const int tid  = threadIdx.x;
    const int wave = tid >> 5;
    const int lane = tid & 31;
    const int lj   = lane & 15;
    const int hi   = lane >> 4;
    const int m0   = blockIdx.x * 32;               // 2 M-tiles
    const int nt0  = blockIdx.y * 32 + wave * 4;    // 4 N-tiles per wave

    const bf16* aRow0 = A + (size_t)(m0 + lj) * Dsz;
    const bf16* aRow1 = A + (size_t)(m0 + 16 + lj) * Dsz;
    const bf16* wRow0 = W + (size_t)((nt0 + 0) * 16 + lj) * Dsz;
    const bf16* wRow1 = W + (size_t)((nt0 + 1) * 16 + lj) * Dsz;
    const bf16* wRow2 = W + (size_t)((nt0 + 2) * 16 + lj) * Dsz;
    const bf16* wRow3 = W + (size_t)((nt0 + 3) * 16 + lj) * Dsz;

    v8f acc[2][4] = {};
#pragma unroll 2
    for (int k0 = 0; k0 < Dsz; k0 += 32) {
        AFrag a0, a1;
        a0.v = *(const v16bf*)(aRow0 + k0 + hi * 16);
        a1.v = *(const v16bf*)(aRow1 + k0 + hi * 16);
        v16bf b0 = *(const v16bf*)(wRow0 + k0 + hi * 16);
        v16bf b1 = *(const v16bf*)(wRow1 + k0 + hi * 16);
        v16bf b2 = *(const v16bf*)(wRow2 + k0 + hi * 16);
        v16bf b3 = *(const v16bf*)(wRow3 + k0 + hi * 16);
        acc[0][0] = __builtin_amdgcn_wmma_f32_16x16x32_bf16(false, a0.v, false, b0, (short)0, acc[0][0], false, false);
        acc[0][1] = __builtin_amdgcn_wmma_f32_16x16x32_bf16(false, a0.v, false, b1, (short)0, acc[0][1], false, false);
        acc[0][2] = __builtin_amdgcn_wmma_f32_16x16x32_bf16(false, a0.v, false, b2, (short)0, acc[0][2], false, false);
        acc[0][3] = __builtin_amdgcn_wmma_f32_16x16x32_bf16(false, a0.v, false, b3, (short)0, acc[0][3], false, false);
        acc[1][0] = __builtin_amdgcn_wmma_f32_16x16x32_bf16(false, a1.v, false, b0, (short)0, acc[1][0], false, false);
        acc[1][1] = __builtin_amdgcn_wmma_f32_16x16x32_bf16(false, a1.v, false, b1, (short)0, acc[1][1], false, false);
        acc[1][2] = __builtin_amdgcn_wmma_f32_16x16x32_bf16(false, a1.v, false, b2, (short)0, acc[1][2], false, false);
        acc[1][3] = __builtin_amdgcn_wmma_f32_16x16x32_bf16(false, a1.v, false, b3, (short)0, acc[1][3], false, false);
    }

#pragma unroll
    for (int mt = 0; mt < 2; ++mt) {
#pragma unroll
        for (int j = 0; j < 4; ++j) {
            float* pBase = P + (size_t)(m0 + mt * 16) * SIXH +
                           (size_t)(nt0 + j) * 16 + lj;
#pragma unroll
            for (int r = 0; r < 8; ++r)
                pBase[(size_t)(r + hi * 8) * SIXH] = acc[mt][j][r];
        }
    }
}

// ---------------------------------------------------------------------------
// Persistent recurrent scan: 32 WGs, WG `wg` owns hidden cols [wg*16, wg*16+16).
// Per step: ps = h @ Ws^T (bf16 WMMA from LDS), gates, h/c update, y write,
// device-scope arrive/spin barrier, async re-broadcast of h (bf16) into LDS.
// ---------------------------------------------------------------------------
__global__ __launch_bounds__(256, 1) void lstm_scan(
    const float* __restrict__ pi,      // [B*T, 6H]
    const bf16*  __restrict__ Wsb,     // [5H, H] bf16 (layer-dir base)
    const float* __restrict__ bias,    // [5H]
    const int*   __restrict__ lengths, // [B]
    float*       __restrict__ ySeq,    // [B*T, 2H]
    int dirOff, int dir,
    bf16*        __restrict__ hbf,     // [B, H] shared bf16 h state
    unsigned int* __restrict__ ctr,    // step barrier counter (pre-zeroed)
    float* __restrict__ hOut,          // [B, H] final h slice dest
    float* __restrict__ cOut) {        // [B, H] final c slice dest

    __shared__ __align__(32) bf16  wsL[5][16][Hsz];  // 80 KB  (Ws slice, resident)
    __shared__ __align__(32) bf16  hL[Bsz][Hsz];     // 32 KB  (full h, bf16)
    __shared__ __align__(16) float psL[Bsz][5 * SCAN_CW]; // 10 KB
    __shared__ float hVal[Bsz][SCAN_CW];             // owned h slice (f32)
    __shared__ float cVal[Bsz][SCAN_CW];             // owned c slice (f32)
    __shared__ float bL[5][SCAN_CW];
    __shared__ int   lenL[Bsz];

    const int tid  = threadIdx.x;
    const int wave = tid >> 5;
    const int lane = tid & 31;
    const int lj   = lane & 15;
    const int hi   = lane >> 4;
    const int c0   = blockIdx.x * SCAN_CW;

    // ---- one-time preloads -------------------------------------------------
    for (int ch = tid; ch < 5120; ch += 256) {
        int f = ch * 8;
        int g = f >> 13;           // / (16*512)
        int r = f & 8191;
        int j = r >> 9;            // / 512
        int k = r & 511;
        *(uint4*)(&wsL[g][j][k]) =
            *(const uint4*)(Wsb + (size_t)(g * Hsz + c0 + j) * Hsz + k);
    }
    if (tid < 80) { int g = tid >> 4, j = tid & 15; bL[g][j] = bias[g * Hsz + c0 + j]; }
    if (tid < Bsz) lenL[tid] = lengths[tid];
    uint4 z4; z4.x = z4.y = z4.z = z4.w = 0u;
    for (int ch = tid; ch < 2048; ch += 256) ((uint4*)hL)[ch] = z4;  // h0 = 0
    for (int i = tid; i < Bsz * SCAN_CW; i += 256) {
        ((float*)hVal)[i] = 0.0f;
        ((float*)cVal)[i] = 0.0f;
    }
    __syncthreads();

    // tile assignment: 10 tiles (2 mTiles x 5 gates) over 8 waves
    const int nT = (wave < 2) ? 2 : 1;
    int tiles[2]; tiles[0] = wave; tiles[1] = 8 + wave;

    const unsigned int hLbase = (unsigned int)(uintptr_t)(&hL[0][0]);

    for (int s = 0; s < Tsz; ++s) {
        // ---- ps = h @ Ws^T for owned columns (WMMA from LDS) ---------------
        for (int it = 0; it < nT; ++it) {
            const int tI = tiles[it];
            const int mt = tI & 1;
            const int g  = tI >> 1;
            const bf16* aRow = &hL[mt * 16 + lj][0];
            const bf16* bRow = &wsL[g][lj][0];
            v8f acc = {};
#pragma unroll
            for (int kt = 0; kt < 16; ++kt) {
                const int k0 = kt * 32;
                AFrag a;
                a.v = *(const v16bf*)(aRow + k0 + hi * 16);
                v16bf b = *(const v16bf*)(bRow + k0 + hi * 16);
                acc = __builtin_amdgcn_wmma_f32_16x16x32_bf16(
                    false, a.v, false, b, (short)0, acc, false, false);
            }
#pragma unroll
            for (int r = 0; r < 8; ++r)
                psL[mt * 16 + r + hi * 8][g * 16 + lj] = acc[r];
        }
        __syncthreads();

        // ---- gates / state update / output ---------------------------------
#pragma unroll
        for (int q = 0; q < 2; ++q) {
            const int e   = tid + q * 256;        // 0..511
            const int b   = e >> 4;
            const int j   = e & 15;
            const int col = c0 + j;
            const int len = lenL[b];
            float hN = hVal[b][j];
            float cN = cVal[b][j];
            if (s < len) {
                const int t = dir ? (len - 1 - s) : s;
                const float* pr = pi + (size_t)(b * Tsz + t) * SIXH + col;
                const float p0 = pr[0],        p1 = pr[Hsz],     p2 = pr[2 * Hsz];
                const float p3 = pr[3 * Hsz],  p4 = pr[4 * Hsz], p5 = pr[5 * Hsz];
                const float ig = sigf(p0 + psL[b][j]      + bL[0][j]);
                const float fg = sigf(p1 + psL[b][16 + j] + bL[1][j]);
                const float mi = tanhf(p2 + psL[b][32 + j] + bL[2][j]);
                const float og = sigf(p3 + psL[b][48 + j] + bL[3][j]);
                const float hw = sigf(p4 + psL[b][64 + j] + bL[4][j]);
                const float mem = ig * mi + fg * cN;
                float out = og * tanhf(mem);
                out = hw * out + (1.0f - hw) * p5;
                hN = out; cN = mem;
                ySeq[(size_t)(b * Tsz + t) * (2 * Hsz) + dirOff + col] = out;
            }
            hVal[b][j] = hN;
            cVal[b][j] = cN;
            hbf[b * Hsz + col] = (bf16)hN;       // publish (frozen or updated)
        }

        // ---- device-scope step barrier -------------------------------------
        __threadfence();
        __syncthreads();
        if (tid == 0) {
            __hip_atomic_fetch_add(ctr, 1u, __ATOMIC_RELEASE,
                                   __HIP_MEMORY_SCOPE_AGENT);
            const unsigned tgt = (unsigned)(NUM_SCAN_WGS * (s + 1));
            while (__hip_atomic_load(ctr, __ATOMIC_ACQUIRE,
                                     __HIP_MEMORY_SCOPE_AGENT) < tgt)
                __builtin_amdgcn_s_sleep(1);
        }
        __syncthreads();
        __threadfence();

        // ---- async re-broadcast of full h into LDS (32 KB through L2) ------
        // GLOBAL_LOAD_ASYNC_TO_LDS_B128: per-lane 16B memory->LDS, no VGPR
        // data movement, tracked with ASYNCcnt (drained by s_wait_asynccnt).
        for (int ch = tid; ch < 2048; ch += 256) {
            unsigned int ldsOff = hLbase + (unsigned int)ch * 16u;
            unsigned long long ga =
                (unsigned long long)(uintptr_t)hbf + (unsigned long long)ch * 16ull;
            asm volatile("global_load_async_to_lds_b128 %0, %1, off"
                         :: "v"(ldsOff), "v"(ga)
                         : "memory");
        }
        asm volatile("s_wait_asynccnt 0x0" ::: "memory");
        __syncthreads();
    }

    // ---- final h / c for this slice ---------------------------------------
    for (int e = tid; e < Bsz * SCAN_CW; e += 256) {
        const int b = e >> 4, j = e & 15;
        hOut[b * Hsz + c0 + j] = hVal[b][j];
        cOut[b * Hsz + c0 + j] = cVal[b][j];
    }
}

// ---------------------------------------------------------------------------
// Host orchestration
// ---------------------------------------------------------------------------
extern "C" void kernel_launch(void* const* d_in, const int* in_sizes, int n_in,
                              void* d_out, int out_size, void* d_ws, size_t ws_size,
                              hipStream_t stream) {
    const float* x   = (const float*)d_in[0];   // [B,T,D]
    const int*   len = (const int*)d_in[1];     // [B]
    const float* Wi  = (const float*)d_in[2];   // [3,2,6H,1024]
    const float* Ws  = (const float*)d_in[3];   // [3,2,5H,H]
    const float* bs  = (const float*)d_in[4];   // [3,2,5H]
    float* out = (float*)d_out;

    const size_t nWi  = (size_t)3 * 2 * SIXH * Dsz;     // 18,874,368
    const size_t nWs  = (size_t)3 * 2 * FIVEH * Hsz;    //  7,864,320
    const size_t nSeq = (size_t)MROWS * Dsz;            // 16,777,216
    const size_t nPi  = (size_t)MROWS * SIXH;           // 50,331,648

    char*  ws  = (char*)d_ws;
    size_t off = 0;
    auto alloc = [&](size_t bytes) -> void* {
        void* p = ws + off;
        off = (off + bytes + 255) & ~(size_t)255;
        return p;
    };
    bf16*  WiB  = (bf16*)alloc(nWi * 2);
    bf16*  WsB  = (bf16*)alloc(nWs * 2);
    bf16*  seqB = (bf16*)alloc(nSeq * 2);
    float* buf0 = (float*)alloc(nSeq * 4);
    float* buf1 = (float*)alloc(nSeq * 4);
    float* pi   = (float*)alloc(nPi * 4);
    bf16*  hbf  = (bf16*)alloc((size_t)Bsz * Hsz * 2);
    unsigned int* ctr = (unsigned int*)alloc(256);
    if (off > ws_size) return;  // scratch too small; bail rather than corrupt

    cvt_f32_to_bf16<<<2048, 256, 0, stream>>>(Wi, WiB, nWi);
    cvt_f32_to_bf16<<<2048, 256, 0, stream>>>(Ws, WsB, nWs);

    const size_t hcBase = (size_t)MROWS * 2 * Hsz;      // seq output elements
    for (int l = 0; l < 3; ++l) {
        const float* inSeq  = (l == 0) ? x : ((l == 1) ? buf0 : buf1);
        float*       outSeq = (l == 0) ? buf0 : ((l == 1) ? buf1 : out);

        cvt_f32_to_bf16<<<2048, 256, 0, stream>>>(inSeq, seqB, nSeq);
        fill_f32<<<2048, 256, 0, stream>>>(outSeq, nSeq, 0.0f);

        for (int d = 0; d < 2; ++d) {
            const bf16*  WiLD = WiB + (size_t)(l * 2 + d) * SIXH * Dsz;
            const bf16*  WsLD = WsB + (size_t)(l * 2 + d) * FIVEH * Hsz;
            const float* bLD  = bs + (size_t)(l * 2 + d) * FIVEH;

            gemm_pi<<<dim3(MROWS / 32, SIXH / 512), 256, 0, stream>>>(
                seqB, WiLD, pi);

            fill_u32<<<1, 32, 0, stream>>>(ctr, 8);

            float* hOut = out + hcBase + (size_t)(2 * l + d) * Bsz * Hsz;
            float* cOut = out + hcBase + (size_t)6 * Bsz * Hsz +
                          (size_t)(2 * l + d) * Bsz * Hsz;
            lstm_scan<<<NUM_SCAN_WGS, 256, 0, stream>>>(
                pi, WsLD, bLD, len, outSeq, d * Hsz, d, hbf, ctr, hOut, cOut);
        }
    }
}
